// SIRnet_63685775065682
// MI455X (gfx1250) — matrix operand order repaired and checked by reading, not compiled, not run on hardware
//
#include <hip/hip_runtime.h>
#include <hip/hip_bf16.h>

typedef __attribute__((ext_vector_type(16))) _Float16 v16h;
typedef __attribute__((ext_vector_type(8)))  _Float16 v8h;
typedef __attribute__((ext_vector_type(8)))  float    v8f;

// ---------------------------------------------------------------------------
// Encoder layer 1: h1[j] = relu(b1[j] + sum_i x[i] * W1[i,j]), D_IN=8192, 128 out
// One block per output j, 256-thread reduction over the 8192-long dot product.
// ---------------------------------------------------------------------------
__global__ void __launch_bounds__(256) enc1_kernel(const float* __restrict__ x,
                                                   const float* __restrict__ w1,
                                                   const float* __restrict__ b1,
                                                   float* __restrict__ h1) {
    __shared__ float red[256];
    const int j = blockIdx.x;             // 0..127
    float acc = 0.0f;
    for (int i = threadIdx.x; i < 8192; i += 256)
        acc += x[i] * w1[i * 128 + j];
    red[threadIdx.x] = acc;
    __syncthreads();
    for (int s = 128; s > 0; s >>= 1) {
        if (threadIdx.x < s) red[threadIdx.x] += red[threadIdx.x + s];
        __syncthreads();
    }
    if (threadIdx.x == 0) h1[j] = fmaxf(red[0] + b1[j], 0.0f);
}

// ---------------------------------------------------------------------------
// Encoder layers 2+3 fused (tiny): h1[128] -> h2[64] -> p[2] = (beta, gamma)
// ---------------------------------------------------------------------------
__global__ void enc23_kernel(const float* __restrict__ h1,
                             const float* __restrict__ w2, const float* __restrict__ b2,
                             const float* __restrict__ w3, const float* __restrict__ b3,
                             float* __restrict__ p) {
    __shared__ float h1s[128];
    __shared__ float h2s[64];
    const int tid = threadIdx.x;          // 128 threads
    h1s[tid] = h1[tid];
    __syncthreads();
    if (tid < 64) {
        float acc = b2[tid];
        for (int i = 0; i < 128; ++i) acc += h1s[i] * w2[i * 64 + tid];
        h2s[tid] = fmaxf(acc, 0.0f);
    }
    __syncthreads();
    if (tid < 2) {
        float acc = b3[tid];
        for (int i = 0; i < 64; ++i) acc += h2s[i] * w3[i * 2 + tid];
        p[tid] = acc;
    }
}

// ---------------------------------------------------------------------------
// Sequential SIR dopri5 scan. Inherently serial recurrence -> one lane keeps
// (S,I,R) in registers; only ILP within each step hides VALU latency.
// sol[T,3] written to workspace.
// ---------------------------------------------------------------------------
__global__ void ode_kernel(const float* __restrict__ t, const float* __restrict__ p,
                           float* __restrict__ sol, int T) {
    if (threadIdx.x != 0 || blockIdx.x != 0) return;
    const float beta = p[0], gamma = p[1];
    float S = 0.99f, I = 0.01f, R = 0.0f;
    sol[0] = S; sol[1] = I; sol[2] = R;
    float tprev = t[0];
    for (int n = 1; n < T; ++n) {
        const float tcur = t[n];
        const float h = (tcur - tprev) * 0.25f;   // dt / SUBSTEPS
        tprev = tcur;
        #pragma unroll 1
        for (int s = 0; s < 4; ++s) {
            // f(S,I): dS=-beta*S*I, dI=beta*S*I-gamma*I, dR=gamma*I
            float k1S, k1I, k1R, k2S, k2I, k2R, k3S, k3I, k3R;
            float k4S, k4I, k4R, k5S, k5I, k5R, k6S, k6I, k6R;
            {   float b = beta * S * I, g = gamma * I;
                k1S = -b; k1I = b - g; k1R = g; }
            {   float yS = S + h * 0.2f * k1S;
                float yI = I + h * 0.2f * k1I;
                float b = beta * yS * yI, g = gamma * yI;
                k2S = -b; k2I = b - g; k2R = g; }
            {   float yS = S + h * (3.0f/40.0f * k1S + 9.0f/40.0f * k2S);
                float yI = I + h * (3.0f/40.0f * k1I + 9.0f/40.0f * k2I);
                float b = beta * yS * yI, g = gamma * yI;
                k3S = -b; k3I = b - g; k3R = g; }
            {   float yS = S + h * (44.0f/45.0f*k1S - 56.0f/15.0f*k2S + 32.0f/9.0f*k3S);
                float yI = I + h * (44.0f/45.0f*k1I - 56.0f/15.0f*k2I + 32.0f/9.0f*k3I);
                float b = beta * yS * yI, g = gamma * yI;
                k4S = -b; k4I = b - g; k4R = g; }
            {   float yS = S + h * (19372.0f/6561.0f*k1S - 25360.0f/2187.0f*k2S
                                  + 64448.0f/6561.0f*k3S - 212.0f/729.0f*k4S);
                float yI = I + h * (19372.0f/6561.0f*k1I - 25360.0f/2187.0f*k2I
                                  + 64448.0f/6561.0f*k3I - 212.0f/729.0f*k4I);
                float b = beta * yS * yI, g = gamma * yI;
                k5S = -b; k5I = b - g; k5R = g; }
            {   float yS = S + h * (9017.0f/3168.0f*k1S - 355.0f/33.0f*k2S
                                  + 46732.0f/5247.0f*k3S + 49.0f/176.0f*k4S
                                  - 5103.0f/18656.0f*k5S);
                float yI = I + h * (9017.0f/3168.0f*k1I - 355.0f/33.0f*k2I
                                  + 46732.0f/5247.0f*k3I + 49.0f/176.0f*k4I
                                  - 5103.0f/18656.0f*k5I);
                float b = beta * yS * yI, g = gamma * yI;
                k6S = -b; k6I = b - g; k6R = g; }
            S += h * (35.0f/384.0f*k1S + 500.0f/1113.0f*k3S + 125.0f/192.0f*k4S
                    - 2187.0f/6784.0f*k5S + 11.0f/84.0f*k6S);
            I += h * (35.0f/384.0f*k1I + 500.0f/1113.0f*k3I + 125.0f/192.0f*k4I
                    - 2187.0f/6784.0f*k5I + 11.0f/84.0f*k6I);
            R += h * (35.0f/384.0f*k1R + 500.0f/1113.0f*k3R + 125.0f/192.0f*k4R
                    - 2187.0f/6784.0f*k5R + 11.0f/84.0f*k6R);
        }
        sol[3*n + 0] = S; sol[3*n + 1] = I; sol[3*n + 2] = R;
    }
}

// ---------------------------------------------------------------------------
// Decoder layers 1+2 fused: sol[T,3] -> relu -> [T,64] -> relu -> A[T,128] f16.
// Weights (3x64 + 64x128) live in LDS; one row per thread; h[64] in registers.
// A is the f16 WMMA A-matrix for the final GEMM.
// ---------------------------------------------------------------------------
__global__ void __launch_bounds__(256) dec12_kernel(const float* __restrict__ sol,
                                                    const float* __restrict__ w1,
                                                    const float* __restrict__ b1,
                                                    const float* __restrict__ w2,
                                                    const float* __restrict__ b2,
                                                    _Float16* __restrict__ A) {
    __shared__ float w1s[192];
    __shared__ float b1s[64];
    __shared__ float w2s[64 * 128];
    __shared__ float b2s[128];
    const int tid = threadIdx.x;
    if (tid < 192) w1s[tid] = w1[tid];
    if (tid < 64)  b1s[tid] = b1[tid];
    for (int i = tid; i < 64 * 128; i += 256) w2s[i] = w2[i];
    if (tid < 128) b2s[tid] = b2[tid];
    __syncthreads();

    const int r = blockIdx.x * 256 + tid;
    const float s0 = sol[3*r], s1 = sol[3*r + 1], s2 = sol[3*r + 2];
    float hreg[64];
    #pragma unroll
    for (int j = 0; j < 64; ++j)
        hreg[j] = fmaxf(b1s[j] + s0*w1s[j] + s1*w1s[64 + j] + s2*w1s[128 + j], 0.0f);
    _Float16* arow = A + (size_t)r * 128;
    for (int j = 0; j < 128; ++j) {
        float acc = b2s[j];
        #pragma unroll
        for (int i = 0; i < 64; ++i) acc += hreg[i] * w2s[i * 128 + j];
        arow[j] = (_Float16)fmaxf(acc, 0.0f);
    }
}

// ---------------------------------------------------------------------------
// Pack dec_w3[128, N] (row-major f32) into the WMMA f16 B-matrix layout:
// per (ntile, kblk) a 32x16 tile; lanes 0-15 hold K=0-15, lanes 16-31 K=16-31
// (contiguous k order, as the A loads use), column = lane&15; 16 contiguous halves per lane.
// ---------------------------------------------------------------------------
__global__ void __launch_bounds__(256) bpack_kernel(const float* __restrict__ w3,
                                                    _Float16* __restrict__ Bp, int N) {
    const int g = blockIdx.x * 256 + threadIdx.x;   // over 128*N, n fastest (coalesced reads)
    const int n = g % N;
    const int k = g / N;
    const int ntile = n >> 4, col = n & 15;
    const int kblk = k >> 5, kk = k & 31;
    const int lane = col + ((kk >> 4) & 1) * 16;
    const int idx  = kk & 15;
    Bp[((((size_t)ntile * 4 + kblk) * 32 + lane) << 4) + idx] = (_Float16)w3[g];
}

// ---------------------------------------------------------------------------
// Final GEMM: out[T,N] = relu(A[T,128] @ W3[128,N] + b3), f16 WMMA, f32 acc.
// One 16x16 output tile per wave; K=128 -> 4 x v_wmma_f32_16x16x32_f16.
// Output-bandwidth bound (256 MB stores @ 23.3 TB/s); A+Bpack are L2-resident.
// grid = (M/128, N/16), 256 threads = 8 waves, wave w -> mtile = bx*8 + w.
// ---------------------------------------------------------------------------
__global__ void __launch_bounds__(256) gemm_kernel(const _Float16* __restrict__ A,
                                                   const _Float16* __restrict__ Bp,
                                                   const float* __restrict__ b3,
                                                   float* __restrict__ out, int N) {
    const int wave  = threadIdx.x >> 5;
    const int lane  = threadIdx.x & 31;
    const int mtile = blockIdx.x * 8 + wave;
    const int ntile = blockIdx.y;

    const int row = lane & 15;               // A-matrix M within tile
    const int kb  = (lane >> 4) << 4;        // 0 or 16 (A K-half select)
    const _Float16* arow  = A + ((size_t)mtile * 16 + row) * 128;
    const v16h*     bbase = (const v16h*)(Bp + (((size_t)ntile * 4) * 32 + lane) * 16);

    v8f c = {};
    #pragma unroll
    for (int kblk = 0; kblk < 4; ++kblk) {
        const int k0 = kblk * 32;
        // A lane data: K = k0+kb .. k0+kb+7 and k0+kb+8 .. k0+kb+15 (two b128 loads)
        v8h lo = *(const v8h*)(arow + k0 + kb);
        v8h hi = *(const v8h*)(arow + k0 + kb + 8);
        v16h a;
        #pragma unroll
        for (int i = 0; i < 8; ++i) { a[i] = lo[i]; a[i + 8] = hi[i]; }
        v16h b = bbase[kblk * 32];           // 32 v16h (= 32 lanes*16 halves) per kblk
        c = __builtin_amdgcn_wmma_f32_16x16x32_f16(
                /*neg_a=*/false, a, /*neg_b=*/false, b,
                /*c_mod=*/(short)0, c, /*reuse_a=*/false, /*reuse_b=*/false);
    }

    const int colg = ntile * 16 + (lane & 15);
    const float bias = b3[colg];
    // C/D layout: VGPR r -> row r (lanes 0-15) / row 8+r (lanes 16-31)
    float* orow = out + ((size_t)mtile * 16 + ((lane >> 4) << 3)) * N + colg;
    #pragma unroll
    for (int r = 0; r < 8; ++r)
        orow[(size_t)r * N] = fmaxf(c[r] + bias, 0.0f);
}

// ---------------------------------------------------------------------------
extern "C" void kernel_launch(void* const* d_in, const int* in_sizes, int n_in,
                              void* d_out, int out_size, void* d_ws, size_t ws_size,
                              hipStream_t stream) {
    const float* x   = (const float*)d_in[0];
    const float* t   = (const float*)d_in[1];
    const float* ew1 = (const float*)d_in[2];
    const float* eb1 = (const float*)d_in[3];
    const float* ew2 = (const float*)d_in[4];
    const float* eb2 = (const float*)d_in[5];
    const float* ew3 = (const float*)d_in[6];
    const float* eb3 = (const float*)d_in[7];
    const float* dw1 = (const float*)d_in[8];
    const float* db1 = (const float*)d_in[9];
    const float* dw2 = (const float*)d_in[10];
    const float* db2 = (const float*)d_in[11];
    const float* dw3 = (const float*)d_in[12];
    const float* db3 = (const float*)d_in[13];
    float* out = (float*)d_out;

    const int T = in_sizes[1];      // 8192 time points (M of the big GEMM)
    const int N = in_sizes[13];     // 8192 output features

    // Workspace layout (all rewritten every call; ~4.2 MB total)
    float* ws_f = (float*)d_ws;
    float*    h1  = ws_f;                                  // 128 f32
    float*    p   = ws_f + 128;                            // 2 f32 (beta, gamma)
    float*    sol = ws_f + 256;                            // T*3 f32
    _Float16* A   = (_Float16*)(ws_f + 256 + (size_t)T*3); // T*128 f16 (32B aligned)
    _Float16* Bp  = A + (size_t)T * 128;                   // 128*N f16 packed B

    enc1_kernel<<<128, 256, 0, stream>>>(x, ew1, eb1, h1);
    enc23_kernel<<<1, 128, 0, stream>>>(h1, ew2, eb2, ew3, eb3, p);
    ode_kernel<<<1, 1, 0, stream>>>(t, p, sol, T);
    dec12_kernel<<<T / 256, 256, 0, stream>>>(sol, dw1, db1, dw2, db2, A);
    bpack_kernel<<<(128 * N) / 256, 256, 0, stream>>>(dw3, Bp, N);
    dim3 grid(T / 128, N / 16);
    gemm_kernel<<<grid, 256, 0, stream>>>(A, Bp, db3, out, N);
}
